// TSSNNHAR_79216376807591
// MI455X (gfx1250) — 16-bit and 8-bit matrix operand order repaired and checked by reading, compiled, not run on hardware
//
#include <hip/hip_runtime.h>

typedef __attribute__((ext_vector_type(16))) __bf16          v16bf;
typedef __attribute__((ext_vector_type(16))) unsigned short  v16us;
typedef __attribute__((ext_vector_type(8)))  float           v8f;
typedef __attribute__((ext_vector_type(8)))  int             v8i;
typedef __attribute__((ext_vector_type(4)))  int             v4i;
typedef __attribute__((ext_vector_type(4)))  unsigned int    v4u;

#define T_STEPS 512
#define BATCH   256
#define IN_DIM  30
#define HID     512
#define NCLS    12
#define BT      16
#define TT      32
#define NCHUNK  (T_STEPS / TT)
#define XCH_ELE (TT * BT * IN_DIM)
#define XCH_B   (XCH_ELE * 4)
#define SROW    576
#define S1_OFF  (2 * XCH_B)
#define SMEM_BYTES (S1_OFF + BT * SROW)

#define W2_SCALE 2873.6953f
#define W2_INV   (1.0f / W2_SCALE)

__device__ __forceinline__ unsigned short f32_bf16(float f) {
  unsigned u = __float_as_uint(f);
  return (unsigned short)((u + 0x7FFFu + ((u >> 16) & 1u)) >> 16);
}

__global__ void pack_w1(const float* __restrict__ W1, unsigned short* __restrict__ out) {
  int g = blockIdx.x * 256 + threadIdx.x;
  if (g >= 32 * 32 * 16) return;
  int j = g & 15, lane = (g >> 4) & 31, nt = g >> 9;
  int k = j + ((lane < 16) ? 0 : 16);
  int n = nt * 16 + (lane & 15);
  out[g] = (k < IN_DIM) ? f32_bf16(W1[n * IN_DIM + k]) : (unsigned short)0;
}

__global__ void pack_w2(const float* __restrict__ W2, unsigned char* __restrict__ out) {
  int g = blockIdx.x * 256 + threadIdx.x;
  if (g >= HID * HID) return;
  int j = g & 31, lane = (g >> 5) & 31, kt = (g >> 10) & 7, nt = g >> 13;
  int k = kt * 64 + ((j < 16) ? (((lane < 16) ? 0 : 16) + j)
                              : (((lane < 16) ? 32 : 48) + (j - 16)));
  int n = nt * 16 + (lane & 15);
  int q = (int)rintf(W2[n * HID + k] * W2_SCALE);
  q = q < -127 ? -127 : (q > 127 ? 127 : q);
  out[g] = (unsigned char)(q & 0xff);
}

__device__ __forceinline__ void tdm_load_x_chunk(const float* xg, int b0, int c,
                                                 unsigned lds_off) {
  unsigned long long ga =
      (unsigned long long)(const void*)(xg + (size_t)b0 * T_STEPS * IN_DIM
                                           + (size_t)c * TT * IN_DIM);
  v4u g0; v8i g1; v4i g2, g3; v8i g4;
  g0[0] = 1u;
  g0[1] = lds_off;
  g0[2] = (unsigned)(ga & 0xffffffffu);
  g0[3] = (unsigned)((ga >> 32) & 0x01ffffffu) | (2u << 30);
  const unsigned td0 = T_STEPS * IN_DIM;
  const unsigned td1 = BATCH;
  const unsigned tl0 = TT * IN_DIM;
  const unsigned tl1 = BT;
  const unsigned st0 = T_STEPS * IN_DIM;
  g1[0] = (int)(2u << 16);
  g1[1] = (int)((td0 & 0xffffu) << 16);
  g1[2] = (int)((td0 >> 16) | ((td1 & 0xffffu) << 16));
  g1[3] = (int)((td1 >> 16) | (tl0 << 16));
  g1[4] = (int)(tl1);
  g1[5] = (int)st0;
  g1[6] = 0;
  g1[7] = 0;
  g2 = (v4i)0; g3 = (v4i)0; g4 = (v8i)0;
  __builtin_amdgcn_tensor_load_to_lds(g0, g1, g2, g3, g4, 0);
}

__global__ __launch_bounds__(512) void snn_fused(
    const float* __restrict__ x,  const float* __restrict__ b1,
    const float* __restrict__ b2, const float* __restrict__ Wh,
    const float* __restrict__ bh, const uint4* __restrict__ w1pk,
    const unsigned char* __restrict__ w2pk, float* __restrict__ out)
{
  extern __shared__ unsigned char smem[];
  unsigned char* s1 = smem + S1_OFF;

  const int tid   = threadIdx.x;
  const int wave  = tid >> 5, lane = tid & 31;
  const int lhalf = lane >> 4, l15 = lane & 15;
  const int b0    = blockIdx.x * BT;

  const int nt0 = wave * 2;
  union U16 { uint4 q[2]; v16us us; v16bf bf; __bf16 h[16]; };
  union U8  { uint4 q[2]; unsigned long long l[4]; v8i i; };

  U16 w1b[2];
  v8i b2r[2][8];
  float b1v[2], b2v[2];
  #pragma unroll
  for (int s = 0; s < 2; ++s) {
    const uint4* p = w1pk + ((nt0 + s) * 32 + lane) * 2;
    w1b[s].q[0] = p[0]; w1b[s].q[1] = p[1];
    int n = (nt0 + s) * 16 + l15;
    b1v[s] = b1[n]; b2v[s] = b2[n];
    #pragma unroll
    for (int kt = 0; kt < 8; ++kt) {
      const uint4* bp =
          (const uint4*)(w2pk + (((size_t)(nt0 + s) * 8 + kt) * 32 + lane) * 32);
      U8 u; u.q[0] = bp[0]; u.q[1] = bp[1];
      b2r[s][kt] = u.i;
    }
  }

  float v1[2][8], v2[2][8], ssum[2][8];
  #pragma unroll
  for (int s = 0; s < 2; ++s)
    #pragma unroll
    for (int r = 0; r < 8; ++r) { v1[s][r] = 0.f; v2[s][r] = 0.f; ssum[s][r] = 0.f; }

  if (wave == 0) {
    tdm_load_x_chunk(x, b0, 0, 0u);
    __builtin_amdgcn_s_wait_tensorcnt(0);
  }
  __syncthreads();

  for (int c = 0; c < NCHUNK; ++c) {
    if (wave == 0 && c + 1 < NCHUNK)
      tdm_load_x_chunk(x, b0, c + 1, (unsigned)(((c + 1) & 1) * XCH_B));
    const float* xb = (const float*)(smem + (c & 1) * XCH_B);

    for (int tt = 0; tt < TT; ++tt) {
      U16 ua;
      #pragma unroll
      for (int j = 0; j < 16; ++j) {
        int k = (lhalf << 4) + j;
        float xv = (k < IN_DIM) ? xb[(l15 * TT + tt) * IN_DIM + k] : 0.0f;
        ua.h[j] = (__bf16)xv;
      }
      #pragma unroll
      for (int s = 0; s < 2; ++s) {
        v8f cz = {0.f,0.f,0.f,0.f,0.f,0.f,0.f,0.f};
        cz = __builtin_amdgcn_wmma_f32_16x16x32_bf16(false, ua.bf, false, w1b[s].bf,
                                                     (short)0, cz, false, false);
        #pragma unroll
        for (int r = 0; r < 8; ++r) {
          float z = cz[r] + b1v[s];
          float v = 0.5f * (v1[s][r] + z);
          bool sp = (v >= 1.0f);
          v1[s][r] = sp ? 0.0f : v;
          int mrow = r + (lhalf << 3);
          s1[mrow * SROW + (nt0 + s) * 16 + l15] = sp ? 0x01 : 0x00;
        }
      }
      __syncthreads();

      v8i acc[2] = { (v8i)0, (v8i)0 };
      #pragma unroll
      for (int kt = 0; kt < 8; ++kt) {
        U8 a8;
        const unsigned char* ap = s1 + l15 * SROW + kt * 64 + (lhalf << 4);
        a8.l[0] = *(const unsigned long long*)(ap);
        a8.l[1] = *(const unsigned long long*)(ap + 8);
        a8.l[2] = *(const unsigned long long*)(ap + 32);
        a8.l[3] = *(const unsigned long long*)(ap + 40);
        acc[0] = __builtin_amdgcn_wmma_i32_16x16x64_iu8(false, a8.i, true, b2r[0][kt],
                                                        acc[0], false, false);
        acc[1] = __builtin_amdgcn_wmma_i32_16x16x64_iu8(false, a8.i, true, b2r[1][kt],
                                                        acc[1], false, false);
      }
      #pragma unroll
      for (int s = 0; s < 2; ++s)
        #pragma unroll
        for (int r = 0; r < 8; ++r) {
          float z = (float)acc[s][r] * W2_INV + b2v[s];
          float v = 0.5f * (v2[s][r] + z);
          bool sp = (v >= 1.0f);
          v2[s][r] = sp ? 0.0f : v;
          ssum[s][r] += sp ? 1.0f : 0.0f;
        }
      __syncthreads();
    }

    if (wave == 0) __builtin_amdgcn_s_wait_tensorcnt(0);
    __syncthreads();
  }

  float* featL = (float*)smem;
  #pragma unroll
  for (int s = 0; s < 2; ++s)
    #pragma unroll
    for (int r = 0; r < 8; ++r)
      featL[(r + (lhalf << 3)) * HID + (nt0 + s) * 16 + l15] =
          ssum[s][r] * (1.0f / (float)T_STEPS);
  __syncthreads();

  if (tid < BT * NCLS) {
    int row = tid / NCLS, cls = tid - row * NCLS;
    float acc = bh[cls];
    for (int h = 0; h < HID; ++h)
      acc += featL[row * HID + h] * Wh[cls * HID + h];
    out[(b0 + row) * NCLS + cls] = acc;
  }
}

extern "C" void kernel_launch(void* const* d_in, const int* in_sizes, int n_in,
                              void* d_out, int out_size, void* d_ws, size_t ws_size,
                              hipStream_t stream) {
  const float* x  = (const float*)d_in[0];
  const float* W1 = (const float*)d_in[1];
  const float* b1 = (const float*)d_in[2];
  const float* W2 = (const float*)d_in[3];
  const float* b2 = (const float*)d_in[4];
  const float* Wh = (const float*)d_in[5];
  const float* bh = (const float*)d_in[6];
  float* out = (float*)d_out;

  unsigned short* w1pk = (unsigned short*)d_ws;
  unsigned char*  w2pk = (unsigned char*)d_ws + 32768;

  pack_w1<<<64, 256, 0, stream>>>(W1, w1pk);
  pack_w2<<<(HID * HID + 255) / 256, 256, 0, stream>>>(W2, w2pk);
  snn_fused<<<BATCH / BT, 512, SMEM_BYTES, stream>>>(
      x, b1, b2, Wh, bh, (const uint4*)w1pk, w2pk, out);
}
